// SoftRank_17300128268842
// MI455X (gfx1250) — compile-verified, matrix operand order checked by reading, not run on hardware
//
#include <hip/hip_runtime.h>
#include <hip/hip_bf16.h>

typedef __attribute__((ext_vector_type(16))) _Float16 v16h;
typedef __attribute__((ext_vector_type(8)))  float    v8f;

#define NN 1024
#define FF 8
#define ALPHA_C 1000.0f
#define LOG2E_C 1.4426950408889634f

#if __has_builtin(__builtin_amdgcn_tanhf)
#define USE_TANH 1
#else
#define USE_TANH 0
#endif

// out[b,j,f] = (1/N) * sum_i sigmoid(ALPHA*(x[b,j,f]-x[b,i,f]))
// sigmoid(z) = 0.5*(1+tanh(z/2)); the affine part folds through the all-ones
// WMMA B-matrix:  sum_i sigmoid = 0.5*sum_i tanh((A/2)(xj-xi)) + N/2
// => feed raw tanh values to WMMA, apply  out = 0.5 + (0.5/N)*S  in epilogue.
__global__ __launch_bounds__(256) void softrank_wmma_kernel(
    const float* __restrict__ x, float* __restrict__ out) {
  __shared__ float tl[NN];  // pre-scaled vector

  const int tid    = threadIdx.x;
  const int bid    = blockIdx.x;   // 0..1023
  const int bf     = bid >> 3;     // 0..127  (b*8+f)
  const int jchunk = bid & 7;      // 0..7
  const int b      = bf >> 3;      // 0..15
  const int f      = bf & 7;       // 0..7

#if USE_TANH
  const float presc = ALPHA_C * 0.5f;            // z/2 = (A/2)*xj - (A/2)*xi
#else
  const float presc = ALPHA_C * LOG2E_C;         // exp(s_i - s_j) = exp2(t_i - t_j)
#endif
  const float* xb = x + b * (NN * FF) + f;
#pragma unroll
  for (int r = 0; r < 4; ++r) {
    const int i = tid + r * 256;
    tl[i] = xb[i * FF] * presc;
  }
  __syncthreads();

  const int lane  = tid & 31;
  const int wave  = tid >> 5;
  const int half  = lane >> 4;   // which half-wave (selects K sub-range)
  const int m     = lane & 15;   // A-matrix row owned by this lane
  const int jbase = jchunk * 128 + wave * 16;
  const float tj  = tl[jbase + m];

  // B matrix = all ones (layout-independent since every element is identical).
  v16h ones;
#pragma unroll
  for (int t = 0; t < 16; ++t) ones[t] = (_Float16)1.0f;

  v8f c = {};
#pragma unroll 2
  for (int kk = 0; kk < NN; kk += 32) {
    const int base = kk + half * 8;
    v16h a;
#pragma unroll
    for (int t = 0; t < 16; ++t) {
      // 16-bit A layout: this lane covers K = base + (t&7) + 16*(t>=8)
      const int i = base + (t & 7) + ((t >> 3) << 4);
#if USE_TANH
      // one trans op per element: v_tanh_f32
      a[t] = (_Float16)__builtin_amdgcn_tanhf(tj - tl[i]);
#else
      const float e = __builtin_amdgcn_exp2f(tl[i] - tj);
      a[t] = (_Float16)__builtin_amdgcn_rcpf(1.0f + e);
#endif
    }
    // D = A*ones + C : every output column n holds the row sums
    c = __builtin_amdgcn_wmma_f32_16x16x32_f16(
        /*neg_a=*/false, a, /*neg_b=*/false, ones,
        /*c_mod=*/(short)0, c, /*reuse_a=*/false, /*reuse_b=*/false);
  }

  // D layout: VGPR r -> M = r + 8*half ; column N = lane&15. Lanes with N==0 write.
  if (m == 0) {
#pragma unroll
    for (int r = 0; r < 8; ++r) {
      const int j = jbase + half * 8 + r;
#if USE_TANH
      out[b * (NN * FF) + j * FF + f] = 0.5f + c[r] * (0.5f / (float)NN);
#else
      out[b * (NN * FF) + j * FF + f] = c[r] * (1.0f / (float)NN);
#endif
    }
  }
}

extern "C" void kernel_launch(void* const* d_in, const int* in_sizes, int n_in,
                              void* d_out, int out_size, void* d_ws, size_t ws_size,
                              hipStream_t stream) {
  const float* x   = (const float*)d_in[0];
  float*       out = (float*)d_out;
  // 128 (b,f) problems * 8 j-chunks = 1024 blocks; 256 threads = 8 waves each.
  softrank_wmma_kernel<<<dim3(1024), dim3(256), 0, stream>>>(x, out);
}
